// MPEdgeNodeBlock_22325240005364
// MI455X (gfx1250) — compile-verified, matrix operand order checked by reading, not run on hardware
//
#include <hip/hip_runtime.h>

// ---------------------------------------------------------------------------
// MPEdgeNodeBlock for MI455X (gfx1250): bf16 WMMA GEMMs with async W->LDS
// staging + fp32 atomic scatter for the segment sums.
// ---------------------------------------------------------------------------

#define NN   25000     // nodes
#define NE   200000    // edges
#define INF  64        // input features
#define OUTF 64        // output features
#define HF   192       // hidden width (2*OUTF + OUTF)

typedef __bf16 bf16;
typedef __attribute__((ext_vector_type(16))) __bf16 v16bf;
typedef __attribute__((ext_vector_type(8)))  __bf16 v8bf;
typedef __attribute__((ext_vector_type(8)))  float  v8f;
typedef __attribute__((ext_vector_type(4)))  int    v4i;

#if defined(__gfx1250__) && __has_builtin(__builtin_amdgcn_global_load_async_to_lds_b128)
#define USE_ASYNC_LDS 1
#else
#define USE_ASYNC_LDS 0
#endif

__device__ __forceinline__ void async_stage_b128(const bf16* g, bf16* l) {
#if USE_ASYNC_LDS
    // signature (probe-confirmed): (AS1 v4i* src, AS3 v4i* dst, Ii offset, Ii cpol)
    __builtin_amdgcn_global_load_async_to_lds_b128(
        (__attribute__((address_space(1))) v4i*)g,
        (__attribute__((address_space(3))) v4i*)l, 0, 0);
#else
    *(v8bf*)l = *(const v8bf*)g;      // global load + ds_store fallback
#endif
}

__device__ __forceinline__ void wait_async_done() {
#if USE_ASYNC_LDS
#if __has_builtin(__builtin_amdgcn_s_wait_asynccnt)
    __builtin_amdgcn_s_wait_asynccnt(0);
#else
    asm volatile("s_wait_asynccnt 0x0" ::: "memory");
#endif
#endif
}

// ---------------------------------------------------------------------------
// elementwise fp32 -> bf16
// ---------------------------------------------------------------------------
__global__ __launch_bounds__(256) void cvt_f32_bf16(const float* __restrict__ src,
                                                    bf16* __restrict__ dst, long n) {
    long i = (long)blockIdx.x * 256 + threadIdx.x;
    if (i < n) dst[i] = (bf16)src[i];
}

// ---------------------------------------------------------------------------
// Generic WMMA GEMM:  Out[M x N] = act( A[M x K] @ W[N x K]^T + bias )
//   A: fp32 or bf16 row-major, W: bf16 row-major (fan_out x fan_in),
//   Out: fp32 or bf16 row-major.  LEAKY: leaky-ReLU with alpha = *alpha_ptr.
//
// W is staged into LDS once per block (async global->LDS b128), rows padded
// to K+8 elements (400B for K=192).  One wave per 16-row strip; wave preloads
// all K/32 A-fragments into VGPRs, then sweeps column tiles TWO at a time with
// independent accumulator chains so the BF16 WMMA->WMMA hazard window of one
// chain is filled by the other chain's wmma + ds_load_b128 instead of v_nops.
//
// Fragment layouts per CDNA5 ISA 7.12.2 (wave32):
//   A (16x32 bf16):  lane l, m = l&15, half = l>>4;
//       elem e: g=e>>3, i=e&7 -> k = kt + g*16 + half*8 + i
//   B (32x16 bf16):  n = l&15; same k pattern; B[k][n] == W[n][k] (row of W)
//   C/D (16x16 f32): VGPR r -> m = r + half*8, n = l&15
// ---------------------------------------------------------------------------
template<int K, int N, bool A_F32, bool OUT_F32, bool LEAKY>
__global__ __launch_bounds__(256) void gemm_wmma(
    const void* __restrict__ Av, const bf16* __restrict__ W,
    const float* __restrict__ bias, void* __restrict__ Outv,
    int M, const float* __restrict__ alpha_ptr)
{
    constexpr int KT = K / 32;       // k-tiles of 32
    constexpr int NT = N / 16;       // n-tiles of 16 (even: 4 or 12)
    constexpr int KP = K + 8;        // padded LDS row (KP*2 bytes, 16B multiple)

    extern __shared__ char smem[];
    bf16* sW = (bf16*)smem;

    // ---- stage W (N x K) into LDS, padded rows, 16B chunks ----
    {
        constexpr int CPR = K / 8;               // 16B chunks per row
        constexpr int TOT = N * CPR;
        for (int c = threadIdx.x; c < TOT; c += 256) {
            const int r  = c / CPR;
            const int cc = c - r * CPR;
            async_stage_b128(W + (long)r * K + cc * 8, sW + r * KP + cc * 8);
        }
        wait_async_done();
        __syncthreads();
    }

    const int lane = threadIdx.x & 31;
    const int wave = threadIdx.x >> 5;
    const long strip = (long)blockIdx.x * 8 + wave;
    const long row0  = strip * 16;
    if (row0 >= M) return;                       // wave-uniform exit (post-barrier)

    const int  half = lane >> 4;                 // 0 or 1
    const int  l15  = lane & 15;
    const bool full_strip = (row0 + 16 <= (long)M);   // wave-uniform

    // ---- preload A fragments for the whole K extent (row m = row0 + l15) ----
    v16bf af[KT];
    const long arow = row0 + l15;
    if (A_F32) {
        const float* ap = (const float*)Av + arow * K;
        if (full_strip) {
            #pragma unroll
            for (int t = 0; t < KT; ++t) {
                const int kb = t * 32 + half * 8;
                #pragma unroll
                for (int i = 0; i < 8; ++i) {
                    af[t][i]     = (bf16)ap[kb + i];
                    af[t][8 + i] = (bf16)ap[kb + 16 + i];
                }
            }
        } else {
            const bool ok = arow < (long)M;
            #pragma unroll
            for (int t = 0; t < KT; ++t) {
                const int kb = t * 32 + half * 8;
                #pragma unroll
                for (int i = 0; i < 8; ++i) {
                    af[t][i]     = (bf16)(ok ? ap[kb + i]      : 0.0f);
                    af[t][8 + i] = (bf16)(ok ? ap[kb + 16 + i] : 0.0f);
                }
            }
        }
    } else {
        const bf16* ap = (const bf16*)Av + arow * K;
        if (full_strip) {
            #pragma unroll
            for (int t = 0; t < KT; ++t) {
                const int kb = t * 32 + half * 8;
                v8bf lo = *(const v8bf*)(ap + kb);
                v8bf hi = *(const v8bf*)(ap + kb + 16);
                #pragma unroll
                for (int i = 0; i < 8; ++i) { af[t][i] = lo[i]; af[t][8 + i] = hi[i]; }
            }
        } else {
            const bool ok = arow < (long)M;
            #pragma unroll
            for (int t = 0; t < KT; ++t) {
                const int kb = t * 32 + half * 8;
                if (ok) {
                    v8bf lo = *(const v8bf*)(ap + kb);
                    v8bf hi = *(const v8bf*)(ap + kb + 16);
                    #pragma unroll
                    for (int i = 0; i < 8; ++i) { af[t][i] = lo[i]; af[t][8 + i] = hi[i]; }
                } else {
                    #pragma unroll
                    for (int i = 0; i < 16; ++i) af[t][i] = (bf16)0.0f;
                }
            }
        }
    }

    const float alpha = LEAKY ? alpha_ptr[0] : 0.0f;
    const long  rbase = row0 + half * 8;

    // ---- sweep column tiles, two independent accumulator chains at a time ----
    #pragma unroll
    for (int nt = 0; nt < NT; nt += 2) {
        const int col0 = nt * 16 + l15;
        const int col1 = col0 + 16;
        const bf16* wr0 = sW + (long)col0 * KP;
        const bf16* wr1 = sW + (long)col1 * KP;
        v8f acc0 = {};
        v8f acc1 = {};
        #pragma unroll
        for (int t = 0; t < KT; ++t) {
            const int kb = t * 32 + half * 8;
            v8bf lo0 = *(const v8bf*)(wr0 + kb);          // ds_load_b128
            v8bf hi0 = *(const v8bf*)(wr0 + kb + 16);
            v8bf lo1 = *(const v8bf*)(wr1 + kb);
            v8bf hi1 = *(const v8bf*)(wr1 + kb + 16);
            v16bf b0, b1;
            #pragma unroll
            for (int i = 0; i < 8; ++i) {
                b0[i] = lo0[i]; b0[8 + i] = hi0[i];
                b1[i] = lo1[i]; b1[8 + i] = hi1[i];
            }
            acc0 = __builtin_amdgcn_wmma_f32_16x16x32_bf16(
                       false, af[t], false, b0, (short)0, acc0, false, false);
            acc1 = __builtin_amdgcn_wmma_f32_16x16x32_bf16(
                       false, af[t], false, b1, (short)0, acc1, false, false);
        }
        // ---- bias + activation + store (D: row = rbase + r, col) ----
        #pragma unroll
        for (int p = 0; p < 2; ++p) {
            const v8f  acc = p ? acc1 : acc0;
            const int  col = p ? col1 : col0;
            const float bv = bias[col];
            if (OUT_F32) {
                float* op = (float*)Outv + rbase * N + col;
                if (full_strip) {
                    #pragma unroll
                    for (int r = 0; r < 8; ++r) {
                        float v = acc[r] + bv;
                        if (LEAKY) v = (v >= 0.0f) ? v : alpha * v;
                        op[(long)r * N] = v;
                    }
                } else {
                    #pragma unroll
                    for (int r = 0; r < 8; ++r) {
                        if (rbase + r < (long)M) {
                            float v = acc[r] + bv;
                            if (LEAKY) v = (v >= 0.0f) ? v : alpha * v;
                            op[(long)r * N] = v;
                        }
                    }
                }
            } else {
                bf16* op = (bf16*)Outv + rbase * N + col;
                if (full_strip) {
                    #pragma unroll
                    for (int r = 0; r < 8; ++r) {
                        float v = acc[r] + bv;
                        if (LEAKY) v = (v >= 0.0f) ? v : alpha * v;
                        op[(long)r * N] = (bf16)v;
                    }
                } else {
                    #pragma unroll
                    for (int r = 0; r < 8; ++r) {
                        if (rbase + r < (long)M) {
                            float v = acc[r] + bv;
                            if (LEAKY) v = (v >= 0.0f) ? v : alpha * v;
                            op[(long)r * N] = (bf16)v;
                        }
                    }
                }
            }
        }
    }
}

// ---------------------------------------------------------------------------
// Build node-aggregation accumulator X (50000 x 192 fp32):
//   cols 0..63  = nodeLin (bf16 -> f32), cols 64..191 = 0 (scatter targets)
// ---------------------------------------------------------------------------
__global__ __launch_bounds__(256) void node_agg_init(const bf16* __restrict__ nodeLin,
                                                     float* __restrict__ X) {
    long i = (long)blockIdx.x * 256 + threadIdx.x;
    const long total = (long)(2 * NN) * HF;
    if (i >= total) return;
    int  c = (int)(i % HF);
    long r = i / HF;
    X[i] = (c < INF) ? (float)nodeLin[r * INF + c] : 0.0f;
}

// ---------------------------------------------------------------------------
// Segment-sum scatter: for each edge e, feature f, stacked half s (real/imag):
//   X[row[e]][64+f]  += nodeLin[col[e]][f]
//   X[row[e]][128+f] += edgeLin[e][f]
// ---------------------------------------------------------------------------
__global__ __launch_bounds__(256) void edge_scatter(const long long* __restrict__ eidx,
                                                    const bf16* __restrict__ nodeLin,
                                                    const bf16* __restrict__ edgeLin,
                                                    float* __restrict__ X) {
    long i = (long)blockIdx.x * 256 + threadIdx.x;
    if (i >= (long)NE * INF) return;
    const int  f = (int)(i & (INF - 1));
    const long e = i >> 6;
    const long row = eidx[2 * e];
    const long col = eidx[2 * e + 1];
    #pragma unroll
    for (int s = 0; s < 2; ++s) {
        float nv = (float)nodeLin[((long)s * NN + col) * INF + f];
        float ev = (float)edgeLin[((long)s * NE + e)  * INF + f];
        float* xr = X + ((long)s * NN + row) * HF;
        atomicAdd(xr + INF + f,     nv);
        atomicAdd(xr + 2 * INF + f, ev);
    }
}

// ---------------------------------------------------------------------------
// Build edge-MLP input E (200000 x 192 bf16) for half s:
//   [ edgeLin_s | h_s[row] | h_s[col] ]   (h read fp32 from d_out)
// ---------------------------------------------------------------------------
__global__ __launch_bounds__(256) void edge_gather(const long long* __restrict__ eidx,
                                                   const bf16* __restrict__ edgeLin,
                                                   const float* __restrict__ h,
                                                   bf16* __restrict__ E, int s) {
    long i = (long)blockIdx.x * 256 + threadIdx.x;
    if (i >= (long)NE * HF) return;
    const int  c = (int)(i % HF);
    const long e = i / HF;
    bf16 v;
    if (c < INF) {
        v = edgeLin[((long)s * NE + e) * INF + c];
    } else {
        const long nidx = (c < 2 * INF) ? eidx[2 * e] : eidx[2 * e + 1];
        const int  f    = (c < 2 * INF) ? (c - INF) : (c - 2 * INF);
        v = (bf16)h[((long)s * NN + nidx) * OUTF + f];
    }
    E[i] = v;
}

// ---------------------------------------------------------------------------
// Host-side orchestration
// ---------------------------------------------------------------------------
static inline int egrid(long n) { return (int)((n + 255) / 256); }
static inline int ggrid(int m)  { return (m + 127) / 128; }       // 8 waves x 16 rows
static inline size_t lds_bytes(int K, int N) { return (size_t)(K + 8) * N * 2; }

extern "C" void kernel_launch(void* const* d_in, const int* in_sizes, int n_in,
                              void* d_out, int out_size, void* d_ws, size_t ws_size,
                              hipStream_t stream) {
    // inputs (setup_inputs dict order)
    const float*     node_real = (const float*)d_in[0];
    const float*     node_imag = (const float*)d_in[1];
    const float*     edge_real = (const float*)d_in[2];
    const float*     edge_imag = (const float*)d_in[3];
    const long long* eidx      = (const long long*)d_in[4];
    const float*     Wn        = (const float*)d_in[5];
    const float*     bn        = (const float*)d_in[6];
    const float*     We        = (const float*)d_in[7];
    const float*     be        = (const float*)d_in[8];
    const float*     node_W    = (const float*)d_in[9];
    const float*     node_b    = (const float*)d_in[10];
    const float*     node_al   = (const float*)d_in[11];
    const float*     node_oW   = (const float*)d_in[12];
    const float*     node_ob   = (const float*)d_in[13];
    const float*     edge_W    = (const float*)d_in[14];
    const float*     edge_b    = (const float*)d_in[15];
    const float*     edge_al   = (const float*)d_in[16];
    const float*     edge_oW   = (const float*)d_in[17];
    const float*     edge_ob   = (const float*)d_in[18];
    (void)in_sizes; (void)n_in; (void)out_size; (void)ws_size;

    // ---- workspace layout (256B aligned slabs) ----
    char*  ws = (char*)d_ws;
    size_t o  = 0;
    auto slab = [&](size_t bytes) -> void* {
        void* p = ws + o;
        o = (o + bytes + 255) & ~(size_t)255;
        return p;
    };
    bf16*  Wn_bf   = (bf16*) slab((size_t)OUTF * INF * 2);
    bf16*  We_bf   = (bf16*) slab((size_t)OUTF * INF * 2);
    bf16*  nW_bf   = (bf16*) slab((size_t)3 * HF * HF * 2);
    bf16*  noW_bf  = (bf16*) slab((size_t)OUTF * HF * 2);
    bf16*  eW_bf   = (bf16*) slab((size_t)3 * HF * HF * 2);
    bf16*  eoW_bf  = (bf16*) slab((size_t)OUTF * HF * 2);
    bf16*  nodeLin = (bf16*) slab((size_t)2 * NN * INF * 2);   // [real;imag] x 64
    bf16*  edgeLin = (bf16*) slab((size_t)2 * NE * INF * 2);
    float* Xf32    = (float*)slab((size_t)2 * NN * HF * 4);    // agg accumulator
    bf16*  Nbuf0   = (bf16*) slab((size_t)2 * NN * HF * 2);    // node MLP ping
    bf16*  Nbuf1   = (bf16*) slab((size_t)2 * NN * HF * 2);    // node MLP pong
    bf16*  Ebuf0   = (bf16*) slab((size_t)NE * HF * 2);        // edge MLP ping
    bf16*  Ebuf1   = (bf16*) slab((size_t)NE * HF * 2);        // edge MLP pong

    float* out_f  = (float*)d_out;                 // [hr | hi | outr | outi]
    float* h_base = out_f;                         // 50000 x 64 (hr stacked hi)

    const size_t lds_in  = lds_bytes(INF, OUTF);   // input linears
    const size_t lds_hid = lds_bytes(HF,  HF);     // hidden layers (76.8 KB)
    const size_t lds_out = lds_bytes(HF,  OUTF);   // output layers

    // ---- 1) weights -> bf16 ----
    cvt_f32_bf16<<<egrid(OUTF*INF),     256, 0, stream>>>(Wn,      Wn_bf,  OUTF*INF);
    cvt_f32_bf16<<<egrid(OUTF*INF),     256, 0, stream>>>(We,      We_bf,  OUTF*INF);
    cvt_f32_bf16<<<egrid(3L*HF*HF),     256, 0, stream>>>(node_W,  nW_bf,  3L*HF*HF);
    cvt_f32_bf16<<<egrid((long)OUTF*HF),256, 0, stream>>>(node_oW, noW_bf, (long)OUTF*HF);
    cvt_f32_bf16<<<egrid(3L*HF*HF),     256, 0, stream>>>(edge_W,  eW_bf,  3L*HF*HF);
    cvt_f32_bf16<<<egrid((long)OUTF*HF),256, 0, stream>>>(edge_oW, eoW_bf, (long)OUTF*HF);

    // ---- 2) input linears (fp32 A -> bf16 out) ----
    gemm_wmma<INF, OUTF, true, false, false><<<ggrid(NN), 256, lds_in, stream>>>(
        node_real, Wn_bf, bn, nodeLin,                     NN, nullptr);
    gemm_wmma<INF, OUTF, true, false, false><<<ggrid(NN), 256, lds_in, stream>>>(
        node_imag, Wn_bf, bn, nodeLin + (long)NN * OUTF,   NN, nullptr);
    gemm_wmma<INF, OUTF, true, false, false><<<ggrid(NE), 256, lds_in, stream>>>(
        edge_real, We_bf, be, edgeLin,                     NE, nullptr);
    gemm_wmma<INF, OUTF, true, false, false><<<ggrid(NE), 256, lds_in, stream>>>(
        edge_imag, We_bf, be, edgeLin + (long)NE * OUTF,   NE, nullptr);

    // ---- 3) node aggregation: init + fp32 atomic segment sums -> bf16 ----
    node_agg_init<<<egrid((long)2 * NN * HF), 256, 0, stream>>>(nodeLin, Xf32);
    edge_scatter <<<egrid((long)NE * INF),    256, 0, stream>>>(eidx, nodeLin, edgeLin, Xf32);
    cvt_f32_bf16 <<<egrid((long)2 * NN * HF), 256, 0, stream>>>(Xf32, Nbuf0, (long)2 * NN * HF);

    // ---- 4) node MLP (3 x 192x192 leaky-ReLU, then 192->64 fp32 into d_out) ----
    gemm_wmma<HF, HF, false, false, true><<<ggrid(2 * NN), 256, lds_hid, stream>>>(
        Nbuf0, nW_bf + 0L * HF * HF, node_b + 0 * HF, Nbuf1, 2 * NN, node_al + 0);
    gemm_wmma<HF, HF, false, false, true><<<ggrid(2 * NN), 256, lds_hid, stream>>>(
        Nbuf1, nW_bf + 1L * HF * HF, node_b + 1 * HF, Nbuf0, 2 * NN, node_al + 1);
    gemm_wmma<HF, HF, false, false, true><<<ggrid(2 * NN), 256, lds_hid, stream>>>(
        Nbuf0, nW_bf + 2L * HF * HF, node_b + 2 * HF, Nbuf1, 2 * NN, node_al + 2);
    gemm_wmma<HF, OUTF, false, true, false><<<ggrid(2 * NN), 256, lds_out, stream>>>(
        Nbuf1, noW_bf, node_ob, h_base, 2 * NN, nullptr);     // -> hr | hi

    // ---- 5) edge MLPs, real then imag (caps workspace) ----
    for (int s = 0; s < 2; ++s) {
        edge_gather<<<egrid((long)NE * HF), 256, 0, stream>>>(eidx, edgeLin, h_base, Ebuf0, s);
        gemm_wmma<HF, HF, false, false, true><<<ggrid(NE), 256, lds_hid, stream>>>(
            Ebuf0, eW_bf + 0L * HF * HF, edge_b + 0 * HF, Ebuf1, NE, edge_al + 0);
        gemm_wmma<HF, HF, false, false, true><<<ggrid(NE), 256, lds_hid, stream>>>(
            Ebuf1, eW_bf + 1L * HF * HF, edge_b + 1 * HF, Ebuf0, NE, edge_al + 1);
        gemm_wmma<HF, HF, false, false, true><<<ggrid(NE), 256, lds_hid, stream>>>(
            Ebuf0, eW_bf + 2L * HF * HF, edge_b + 2 * HF, Ebuf1, NE, edge_al + 2);
        gemm_wmma<HF, OUTF, false, true, false><<<ggrid(NE), 256, lds_out, stream>>>(
            Ebuf1, eoW_bf, edge_ob,
            out_f + (long)2 * NN * OUTF + (long)s * NE * OUTF, NE, nullptr);
    }
}
